// Involution2d_73675868996109
// MI455X (gfx1250) — matrix operand order repaired and checked by reading, not compiled, not run on hardware
//
#include <hip/hip_runtime.h>

// ---------------- CDNA5 WMMA types ----------------
typedef __bf16 bf16_t;
typedef __attribute__((ext_vector_type(16))) __bf16 v16bf;
typedef __attribute__((ext_vector_type(8)))  float  v8f;
typedef __attribute__((__vector_size__(4 * sizeof(int)))) int v4i;

#define B_    16
#define C_    256
#define CR    64
#define H_    64
#define W_    64
#define G_    16
#define KS    7
#define PADL  3
#define KK    49
#define MROWS 784           // K*K*G rows of w_span
#define EPS   1e-5f

// gfx1250 async global->LDS DMA path (guarded; falls back to plain loads)
#if __has_builtin(__builtin_amdgcn_global_load_async_to_lds_b128) && \
    __has_builtin(__builtin_amdgcn_s_wait_asynccnt)
#define HAVE_ASYNC_LDS 1
typedef __attribute__((address_space(1))) v4i* gptr_b128;
typedef __attribute__((address_space(3))) v4i* lptr_b128;
#else
#define HAVE_ASYNC_LDS 0
#endif

static __device__ __forceinline__ bf16_t tobf(float f) { return (bf16_t)f; }

// LDS layout (dynamic):
//   sx     : C_*W_   bf16  (x row, GEMM1 B operand)      32 KB
//   sr     : CR*W_   bf16  (BN+ReLU'd reduce output)      8 KB
//   sw     : MROWS*W_ bf16 (per-pixel dynamic kernels)  100 KB
//            (first 64 KB double as fp32 staging for the async x-row DMA)
//   sscale : CR float, sshift : CR float                 0.5 KB
#define SMEM_BYTES ((C_*W_ + CR*W_ + MROWS*W_) * 2 + 2 * CR * 4)

__global__ __launch_bounds__(256)
void involution_fused_kernel(const float* __restrict__ x,
                             const float* __restrict__ w_reduce,
                             const float* __restrict__ w_span,
                             const float* __restrict__ bn_gamma,
                             const float* __restrict__ bn_beta,
                             const float* __restrict__ bn_mean,
                             const float* __restrict__ bn_var,
                             float* __restrict__ out) {
  extern __shared__ __attribute__((aligned(32))) char smem_raw[];
  bf16_t* sx = (bf16_t*)smem_raw;           // [C_][W_]
  bf16_t* sr = sx + C_ * W_;                // [CR][W_]
  bf16_t* sw = sr + CR * W_;                // [MROWS][W_]
  float* sscale = (float*)(sw + MROWS * W_);
  float* sshift = sscale + CR;

  const int blk  = blockIdx.x;
  const int b    = blk >> 6;                 // batch
  const int h    = blk & 63;                 // row
  const int tid  = threadIdx.x;
  const int lane = tid & 31;
  const int wave = tid >> 5;

  // ---------- Stage 0: x row -> LDS (bf16, coalesced); BN scale/shift ----------
  // float4 index q = it*256 + tid  ->  ch = q>>4, px4 = q&15 (lane-contiguous)
#if HAVE_ASYNC_LDS
  {
    float* stage = (float*)sw;   // 64 KB fp32 staging inside the sw region
#pragma unroll
    for (int it = 0; it < 16; ++it) {
      const int q   = it * 256 + tid;
      const int chq = q >> 4, px4 = q & 15;
      const float* g = x + (((size_t)b * C_ + chq) * H_ + h) * W_ + px4 * 4;
      __builtin_amdgcn_global_load_async_to_lds_b128(
          (gptr_b128)g, (lptr_b128)(stage + q * 4), 0, 0);
    }
    __builtin_amdgcn_s_wait_asynccnt(0);
    __syncthreads();
#pragma unroll
    for (int it = 0; it < 16; ++it) {
      const int q   = it * 256 + tid;
      const int chq = q >> 4, px4 = q & 15;
      float4 v = ((const float4*)stage)[q];
      bf16_t* d = sx + chq * W_ + px4 * 4;
      d[0] = tobf(v.x); d[1] = tobf(v.y); d[2] = tobf(v.z); d[3] = tobf(v.w);
    }
  }
#else
  {
#pragma unroll
    for (int it = 0; it < 16; ++it) {
      const int q   = it * 256 + tid;
      const int chq = q >> 4, px4 = q & 15;
      float4 v = *(const float4*)(x + (((size_t)b * C_ + chq) * H_ + h) * W_ + px4 * 4);
      bf16_t* d = sx + chq * W_ + px4 * 4;
      d[0] = tobf(v.x); d[1] = tobf(v.y); d[2] = tobf(v.z); d[3] = tobf(v.w);
    }
  }
#endif
  if (tid < CR) {
    float sc = bn_gamma[tid] * rsqrtf(bn_var[tid] + EPS);
    sscale[tid] = sc;
    sshift[tid] = bn_beta[tid] - bn_mean[tid] * sc;
  }
  __syncthreads();

  // ---------- Stage 1: r = w_reduce (64x256) x x (256 x 64px), WMMA bf16 ----------
  // wave -> M-tile = wave&3, N-tiles {2*(wave>>2), 2*(wave>>2)+1}; A gathered once per K-step.
  {
    const int mt  = wave & 3;
    const int nt0 = (wave >> 2) * 2;
    v8f acc0 = {}, acc1 = {};
    const int m = mt * 16 + (lane & 15);
    for (int kk = 0; kk < 8; ++kk) {
      v16bf a;
      const int kbase = kk * 32 + ((lane >> 4) << 3);
#pragma unroll
      for (int e = 0; e < 16; ++e) {      // A: e<8 -> k=kbase+e ; e>=8 -> k=kbase+16+(e-8)
        const int k = kbase + ((e & 8) << 1) + (e & 7);
        a[e] = tobf(w_reduce[m * C_ + k]);
      }
      v16bf bv0, bv1;
#pragma unroll
      for (int e = 0; e < 16; ++e) {      // B: same k as A, n = lane&15
        const bf16_t* brow = sx + (kbase + ((e & 8) << 1) + (e & 7)) * W_ + nt0 * 16 + (lane & 15);
        bv0[e] = brow[0];
        bv1[e] = brow[16];
      }
      acc0 = __builtin_amdgcn_wmma_f32_16x16x32_bf16(false, a, false, bv0,
                                                     (short)0, acc0, false, false);
      acc1 = __builtin_amdgcn_wmma_f32_16x16x32_bf16(false, a, false, bv1,
                                                     (short)0, acc1, false, false);
    }
    // BN + ReLU fused; D layout: VGPR v -> M = v + 8*(lane>=16), N = lane&15
    const int n0    = nt0 * 16 + (lane & 15);
    const int mbase = mt * 16 + ((lane >> 4) << 3);
#pragma unroll
    for (int v = 0; v < 8; ++v) {
      const int cr = mbase + v;
      const float sc = sscale[cr], sh = sshift[cr];
      sr[cr * W_ + n0]      = tobf(fmaxf(acc0[v] * sc + sh, 0.f));
      sr[cr * W_ + n0 + 16] = tobf(fmaxf(acc1[v] * sc + sh, 0.f));
    }
  }
  __syncthreads();

  // ---------- Stage 2: w = w_span (784x64) x r (64 x 64px), WMMA bf16 ----------
  // wave sweeps M-tiles (49 over 8 waves); A (both K-steps) gathered once, reused for 4 N-tiles.
  for (int mt = wave; mt < 49; mt += 8) {
    v16bf a0, a1;
    const int m = mt * 16 + (lane & 15);
    {
      const int kb0 = ((lane >> 4) << 3);
#pragma unroll
      for (int e = 0; e < 16; ++e) {
        const int k = kb0 + ((e & 8) << 1) + (e & 7);
        a0[e] = tobf(w_span[m * CR + k]);
        a1[e] = tobf(w_span[m * CR + 32 + k]);
      }
    }
    const int n_lo  = lane & 15;
    const int mbase = mt * 16 + ((lane >> 4) << 3);
#pragma unroll
    for (int nt = 0; nt < 4; ++nt) {
      v8f acc = {};
      v16bf bv0, bv1;
#pragma unroll
      for (int e = 0; e < 16; ++e) {
        const int k = ((lane >> 4) << 3) + ((e & 8) << 1) + (e & 7);
        bv0[e] = sr[k * W_ + nt * 16 + n_lo];
        bv1[e] = sr[(32 + k) * W_ + nt * 16 + n_lo];
      }
      acc = __builtin_amdgcn_wmma_f32_16x16x32_bf16(false, a0, false, bv0,
                                                    (short)0, acc, false, false);
      acc = __builtin_amdgcn_wmma_f32_16x16x32_bf16(false, a1, false, bv1,
                                                    (short)0, acc, false, false);
      const int n = nt * 16 + n_lo;
#pragma unroll
      for (int v = 0; v < 8; ++v)
        sw[(mbase + v) * W_ + n] = tobf(acc[v]);
    }
  }
  __syncthreads();

  // ---------- Stage 3: involution (49-tap dynamic conv), x halo from L2 ----------
  // thread = channel; 4 chunks of 16 px with a 22-wide register halo row.
  const int ch = tid;
  const int g  = ch >> 4;
  const float* xch = x + (((size_t)b * C_ + ch) * H_) * W_;
  float* och = out + (((size_t)b * C_ + ch) * H_ + h) * W_;
#pragma unroll
  for (int cb = 0; cb < 4; ++cb) {
    const int pxb = cb * 16;
    float acc[16];
#pragma unroll
    for (int e = 0; e < 16; ++e) acc[e] = 0.f;
    for (int i = 0; i < KS; ++i) {
      const int row = h + i - PADL;
      if (row < 0 || row >= H_) continue;
      float xr[22];
#pragma unroll
      for (int e = 0; e < 22; ++e) {
        const int col = pxb + e - PADL;
        xr[e] = (col >= 0 && col < W_) ? xch[row * W_ + col] : 0.f;
      }
#pragma unroll
      for (int j = 0; j < KS; ++j) {
        const bf16_t* wrow = sw + (g * KK + i * KS + j) * W_ + pxb;
#pragma unroll
        for (int e = 0; e < 16; ++e)
          acc[e] += (float)wrow[e] * xr[e + j];   // x col = pxb+e+j-3
      }
    }
    float4* o4 = (float4*)(och + pxb);
#pragma unroll
    for (int q = 0; q < 4; ++q)
      o4[q] = make_float4(acc[q * 4 + 0], acc[q * 4 + 1], acc[q * 4 + 2], acc[q * 4 + 3]);
  }
}

extern "C" void kernel_launch(void* const* d_in, const int* in_sizes, int n_in,
                              void* d_out, int out_size, void* d_ws, size_t ws_size,
                              hipStream_t stream) {
  (void)in_sizes; (void)n_in; (void)d_ws; (void)ws_size; (void)out_size;
  const float* x        = (const float*)d_in[0];
  const float* w_reduce = (const float*)d_in[1];
  const float* w_span   = (const float*)d_in[2];
  const float* bn_gamma = (const float*)d_in[3];
  const float* bn_beta  = (const float*)d_in[4];
  const float* bn_mean  = (const float*)d_in[5];
  const float* bn_var   = (const float*)d_in[6];
  float* out = (float*)d_out;

  dim3 grid(B_ * H_);   // one workgroup per (batch, row)
  dim3 block(256);      // 8 waves (wave32)
  involution_fused_kernel<<<grid, block, SMEM_BYTES, stream>>>(
      x, w_reduce, w_span, bn_gamma, bn_beta, bn_mean, bn_var, out);
}
